// DependencyEncoder_26147760898495
// MI455X (gfx1250) — matrix operand order repaired and checked by reading, not compiled, not run on hardware
//
#include <hip/hip_runtime.h>
#include <cstdint>
#include <cstddef>

// ---------------------------------------------------------------------------
// SPINN tree-LSTM encoder for MI455X (gfx1250).
// Design: 32 persistent workgroups x 512 threads (16 wave32s). Each WG owns a
// 16-row batch tile (M=16 WMMA tiles) and runs all 125 sequential steps.
// Weights are packed once (pack_kernel) into bf16, transposed [N][K] with
// gate-blocked column permutation; per step they stream from L2 into
// v_wmma_f32_16x16x32_bf16. Activation A-panels live in LDS; per-row
// head/childL/childR selection is done with LDS base-offset indirection
// (rows point into top / sec / zero regions). Stacks live in workspace
// (h: bf16 GEMM feed, c: f32 recurrence).
// ---------------------------------------------------------------------------

typedef __attribute__((ext_vector_type(16))) __bf16 v16bf;
typedef __attribute__((ext_vector_type(8)))  float  v8f;
typedef __attribute__((ext_vector_type(4)))  unsigned int u32x4;

#define Bn   512
#define Ln   64
#define Hn   512
#define TDn  64
#define Tn   125
#define SMAX 66
#define KA   1600     // packed K: buf(512)|top(512)|sec(512)|th(64)
#define NTRK 256
#define NCMP 2560
#define NTHREADS 512  // 16 wave32s
#define NWAVES   16

// ---- workspace layout (bytes) ----
#define WS_WT_TRK   ((size_t)0)
#define WS_WT_CMP   (WS_WT_TRK + (size_t)NTRK*KA*2)             // 819,200
#define WS_BIAS_TRK (WS_WT_CMP + (size_t)NCMP*KA*2)             // +8,192,000
#define WS_BIAS_CMP (WS_BIAS_TRK + (size_t)NTRK*4)
#define WS_STACK_H  (WS_BIAS_CMP + (size_t)NCMP*4)
#define WS_STACK_C  (WS_STACK_H + (size_t)Bn*SMAX*Hn*2)
// total ~112.8 MB, L2-resident working set

// ---- dynamic LDS layout (bytes) ----
#define SM_ASTAGE 0                         // bf16 [16][1600]  (51200 B)
#define SM_ZROW   51200                     // bf16 [512] zeros (1024 B)
#define SM_CST    52224                     // f32  [2][16][512] top/sec c (65536)
#define SM_GATES  117760                    // f32  [16][256] tracker gates (16384)
#define SM_TH     134144                    // f32  [16][64]
#define SM_TC     138240                    // f32  [16][64]
#define SM_BT     142336                    // f32  [256] tracker bias
#define SM_BC     143360                    // f32  [2560] comp bias (permuted)
#define SM_SP     153600                    // int  [16]
#define SM_BP     153664
#define SM_WIDX   153728
#define SM_SHIFT  153792
#define SM_CHEAD  153856
#define SM_SEG    153920                    // int  [16][4] A-seg byte offsets
#define SM_TOTAL  154176

__device__ __forceinline__ float sigf(float x) { return 1.0f / (1.0f + __expf(-x)); }

// ---------------------------------------------------------------------------
// Pack weights: transpose to [N][K] bf16, concatenate K-segments, permute
// composition columns into (wave,group,gate) blocks of 16.
// packed comp row p: bg=p/80, q=(p%80)/16, n=p%16  ->  orig col q*512 + bg*16+n
// ---------------------------------------------------------------------------
__global__ void pack_kernel(const float* __restrict__ W_comp,
                            const float* __restrict__ U_head_w,
                            const float* __restrict__ U_head_b,
                            const float* __restrict__ U_cl,
                            const float* __restrict__ U_cr,
                            const float* __restrict__ W_ih,
                            const float* __restrict__ W_hh,
                            const float* __restrict__ b_ih,
                            const float* __restrict__ b_hh,
                            char* __restrict__ ws) {
  __bf16* WtT = (__bf16*)(ws + WS_WT_TRK);
  __bf16* WtC = (__bf16*)(ws + WS_WT_CMP);
  float*  bT  = (float*)(ws + WS_BIAS_TRK);
  float*  bC  = (float*)(ws + WS_BIAS_CMP);
  const int totT = NTRK * KA, totC = NCMP * KA;
  const int total = totT + totC + NTRK + NCMP;
  for (int i = blockIdx.x * blockDim.x + threadIdx.x; i < total;
       i += gridDim.x * blockDim.x) {
    if (i < totT) {
      int n = i / KA, k = i % KA;
      float v = (k < 3 * Hn) ? W_ih[(size_t)k * NTRK + n]
                             : W_hh[(size_t)(k - 3 * Hn) * NTRK + n];
      WtT[i] = (__bf16)v;
    } else if (i < totT + totC) {
      int j = i - totT;
      int p = j / KA, k = j % KA;
      int bg = p / 80, r = p % 80, q = r / 16, n = r % 16;
      int c = q * Hn + bg * 16 + n;
      float v;
      if      (k <     Hn) v = U_head_w[(size_t)k * NCMP + c];
      else if (k < 2 * Hn) v = U_cl[(size_t)(k -     Hn) * NCMP + c];
      else if (k < 3 * Hn) v = U_cr[(size_t)(k - 2 * Hn) * NCMP + c];
      else                 v = W_comp[(size_t)(k - 3 * Hn) * NCMP + c];
      WtC[j] = (__bf16)v;
    } else if (i < totT + totC + NTRK) {
      int n = i - totT - totC;
      bT[n] = b_ih[n] + b_hh[n];
    } else {
      int p = i - totT - totC - NTRK;
      int bg = p / 80, r = p % 80, q = r / 16, n = r % 16;
      bC[p] = U_head_b[q * Hn + bg * 16 + n];
    }
  }
}

// ---------------------------------------------------------------------------
// Main persistent kernel.
// ---------------------------------------------------------------------------
__global__ __launch_bounds__(NTHREADS)
void spinn_kernel(const float* __restrict__ seq,   // (512,64,1024)
                  const int*   __restrict__ trans, // (512,125)
                  const float* __restrict__ th0,   // (512,64)
                  const float* __restrict__ tc0,   // (512,64)
                  char* __restrict__ ws,
                  float* __restrict__ out) {       // (512,512)
  extern __shared__ char smem[];
  __bf16* Astage = (__bf16*)(smem + SM_ASTAGE);
  __bf16* Zrow   = (__bf16*)(smem + SM_ZROW);
  float*  Cst    = (float*)(smem + SM_CST);
  float*  gatesS = (float*)(smem + SM_GATES);
  float*  thS    = (float*)(smem + SM_TH);
  float*  tcS    = (float*)(smem + SM_TC);
  float*  biasT  = (float*)(smem + SM_BT);
  float*  biasC  = (float*)(smem + SM_BC);
  int* sh_sp    = (int*)(smem + SM_SP);
  int* sh_bp    = (int*)(smem + SM_BP);
  int* sh_widx  = (int*)(smem + SM_WIDX);
  int* sh_shift = (int*)(smem + SM_SHIFT);
  int* sh_cHead = (int*)(smem + SM_CHEAD);
  int* sh_seg   = (int*)(smem + SM_SEG);

  const __bf16* WtT = (const __bf16*)(ws + WS_WT_TRK);
  const __bf16* WtC = (const __bf16*)(ws + WS_WT_CMP);
  const float*  bTg = (const float*)(ws + WS_BIAS_TRK);
  const float*  bCg = (const float*)(ws + WS_BIAS_CMP);
  __bf16* stackH = (__bf16*)(ws + WS_STACK_H);
  float*  stackC = (float*)(ws + WS_STACK_C);

  const int tid  = threadIdx.x;
  const int wg   = blockIdx.x;       // 0..31 -> rows wg*16..+15
  const int wv   = tid >> 5;         // wave 0..15
  const int lane = tid & 31;
  const int nB   = lane & 15;        // M row (A) / N col (B,C,D)
  const int hi   = lane >> 4;        // lane-half

  // ---- prologue: biases, zero row, tracker state, stack init ----
  for (int i = tid; i < NTRK; i += NTHREADS) biasT[i] = bTg[i];
  for (int i = tid; i < NCMP; i += NTHREADS) biasC[i] = bCg[i];
  for (int i = tid; i < 512;  i += NTHREADS) Zrow[i] = (__bf16)0.0f;
  for (int i = tid; i < 16 * TDn; i += NTHREADS) {
    int m = i >> 6, j = i & 63;
    int b = wg * 16 + m;
    float th = th0[(size_t)b * TDn + j];
    float tc = tc0[(size_t)b * TDn + j];
    thS[i] = th; tcS[i] = tc;
    Astage[m * KA + 3 * Hn + j] = (__bf16)th;
  }
  for (int i = tid; i < 16 * Hn; i += NTHREADS) {
    int m = i >> 9, c = i & 511;
    int b = wg * 16 + m;
    float h0v = seq[((size_t)b * Ln + 0) * (2 * Hn) + c];
    float c0v = seq[((size_t)b * Ln + 0) * (2 * Hn) + Hn + c];
    size_t s0 = ((size_t)b * SMAX + 0) * Hn + c;
    size_t s1 = ((size_t)b * SMAX + 1) * Hn + c;
    stackH[s0] = (__bf16)h0v; stackH[s1] = (__bf16)h0v;
    stackC[s0] = c0v;         stackC[s1] = c0v;
  }
  if (tid < 16) { sh_sp[tid] = 2; sh_bp[tid] = 0; }
  __threadfence_block();
  __syncthreads();

  for (int t = 0; t < Tn; ++t) {
    // ---- phase 1: per-row control + stage A-panel / c-values ----
    if (tid < 16) {
      int m = tid, b = wg * 16 + m;
      int tr = trans[(size_t)b * Tn + t];
      int isShift = (tr == 1);
      int isLeft  = (tr == 2);
      int dsp = isShift ? 1 : -1;
      sh_widx[m]  = sh_sp[m] + dsp - 1;
      sh_shift[m] = isShift;
      sh_cHead[m] = isLeft ? 0 : 1;      // head c-value from top(0) or sec(1)
      int rowB = m * (KA * 2);           // 3200 bytes per A row
      // A-panel byte offsets: top @ +1024, sec @ +2048, th @ +3072
      sh_seg[m * 4 + 0] = isLeft ? (rowB + 1024) : (rowB + 2048); // head
      sh_seg[m * 4 + 1] = isLeft ? (rowB + 2048) : SM_ZROW;       // childL (masked)
      sh_seg[m * 4 + 2] = isLeft ? SM_ZROW : (rowB + 1024);       // childR (masked)
      sh_seg[m * 4 + 3] = rowB + 3072;                            // tracker h
    }
    for (int i = tid; i < 16 * Hn; i += NTHREADS) {
      int m = i >> 9, c = i & 511;
      int b = wg * 16 + m;
      int sp = sh_sp[m], bp = sh_bp[m];
      size_t ptop = ((size_t)b * SMAX + (sp - 1)) * Hn + c;
      size_t psec = ((size_t)b * SMAX + (sp - 2)) * Hn + c;
      Astage[m * KA + c]          = (__bf16)seq[((size_t)b * Ln + bp) * (2 * Hn) + c];
      Astage[m * KA + Hn + c]     = stackH[ptop];
      Astage[m * KA + 2 * Hn + c] = stackH[psec];
      Cst[(0 * 16 + m) * 512 + c] = stackC[ptop];
      Cst[(1 * 16 + m) * 512 + c] = stackC[psec];
    }
    __syncthreads();

    // ---- phase 2: tracker GEMM (16x256 <- 16x1600 @ 1600x256) ----
    {
      const int nt = wv;  // one 16-wide N tile per wave
      v8f acc = {0.f,0.f,0.f,0.f,0.f,0.f,0.f,0.f};
      const char* bbase = (const char*)WtT + (size_t)(nt * 16 + nB) * (KA * 2);
      const int arow = nB * (KA * 2);
      for (int kt = 0; kt < 50; ++kt) {
        union { u32x4 q[2]; v16bf v; } av, bv;
        const char* ap = smem + arow + kt * 64 + hi * 16;
        av.q[0] = *(const u32x4*)ap;
        av.q[1] = *(const u32x4*)(ap + 32);
        const char* bp_ = bbase + kt * 64 + hi * 16;
        bv.q[0] = *(const u32x4*)bp_;
        bv.q[1] = *(const u32x4*)(bp_ + 32);
        acc = __builtin_amdgcn_wmma_f32_16x16x32_bf16(
            false, av.v, false, bv.v, (short)0, acc, false, false);
      }
#pragma unroll
      for (int rr = 0; rr < 8; ++rr) {
        int m = rr + hi * 8;
        gatesS[m * NTRK + nt * 16 + nB] = acc[rr];
      }
    }
    __syncthreads();

    // ---- phase 3: tracker LSTM elementwise, refresh th in A-panel ----
    for (int i = tid; i < 16 * TDn; i += NTHREADS) {
      int m = i >> 6, j = i & 63;
      float ig = gatesS[m * NTRK + j]        + biasT[j];
      float fg = gatesS[m * NTRK + 64 + j]   + biasT[64 + j];
      float gg = gatesS[m * NTRK + 128 + j]  + biasT[128 + j];
      float og = gatesS[m * NTRK + 192 + j]  + biasT[192 + j];
      float tc = sigf(fg) * tcS[i] + sigf(ig) * tanhf(gg);
      float th = sigf(og) * tanhf(tc);
      tcS[i] = tc; thS[i] = th;
      Astage[m * KA + 3 * Hn + j] = (__bf16)th;
    }
    __syncthreads();

    // ---- phase 4: composition GEMM (5 gate tiles in regs) + TreeLSTM ----
    for (int g = 0; g < 2; ++g) {
      const int bg = wv * 2 + g;           // 0..31: 16-wide h-column block
      v8f acc[5];
#pragma unroll
      for (int q = 0; q < 5; ++q) acc[q] = (v8f){0.f,0.f,0.f,0.f,0.f,0.f,0.f,0.f};
      const char* bq[5];
#pragma unroll
      for (int q = 0; q < 5; ++q)
        bq[q] = (const char*)WtC + (size_t)((bg * 5 + q) * 16 + nB) * (KA * 2);
      for (int kt = 0; kt < 50; ++kt) {
        int seg = kt >> 4;                 // 0:head 1:childL 2:childR 3:th
        int lk  = kt & 15;
        int aoff = sh_seg[nB * 4 + seg] + lk * 64 + hi * 16;
        union { u32x4 q[2]; v16bf v; } av;
        av.q[0] = *(const u32x4*)(smem + aoff);
        av.q[1] = *(const u32x4*)(smem + aoff + 32);
#pragma unroll
        for (int q = 0; q < 5; ++q) {
          union { u32x4 qq[2]; v16bf v; } bv;
          const char* p = bq[q] + kt * 64 + hi * 16;
          bv.qq[0] = *(const u32x4*)p;
          bv.qq[1] = *(const u32x4*)(p + 32);
          acc[q] = __builtin_amdgcn_wmma_f32_16x16x32_bf16(
              false, av.v, false, bv.v, (short)0, acc[q], false, false);
        }
      }
      const int col = bg * 16 + nB;
      float bi[5];
#pragma unroll
      for (int q = 0; q < 5; ++q) bi[q] = biasC[(bg * 5 + q) * 16 + nB];
#pragma unroll
      for (int rr = 0; rr < 8; ++rr) {
        int m = rr + hi * 8;
        int b = wg * 16 + m;
        float iv = acc[0][rr] + bi[0];
        float ov = acc[1][rr] + bi[1];
        float fh = acc[2][rr] + bi[2];
        float fc = acc[3][rr] + bi[3];
        float uv = acc[4][rr] + bi[4];
        int ch = sh_cHead[m];
        float hc = Cst[(ch * 16 + m) * 512 + col];
        float cc = Cst[((1 - ch) * 16 + m) * 512 + col];
        float cn = sigf(iv) * tanhf(uv) + sigf(fh) * hc + sigf(fc) * cc;
        float hn = sigf(ov) * tanhf(cn);
        float ph, pc;
        if (sh_shift[m]) {
          size_t tb = ((size_t)b * Ln + sh_bp[m]) * (2 * Hn);
          ph = seq[tb + col];
          pc = seq[tb + Hn + col];
        } else { ph = hn; pc = cn; }
        size_t sidx = ((size_t)b * SMAX + sh_widx[m]) * Hn + col;
        stackH[sidx] = (__bf16)ph;
        stackC[sidx] = pc;
        if (t == Tn - 1) out[(size_t)b * Hn + col] = ph;
      }
    }
    __threadfence_block();
    __syncthreads();
    // ---- phase 5: advance per-row stack/buffer pointers ----
    if (tid < 16) {
      int m = tid;
      sh_sp[m] += sh_shift[m] ? 1 : -1;
      int nb = sh_bp[m] + sh_shift[m];
      sh_bp[m] = (nb > Ln - 1) ? (Ln - 1) : nb;
    }
    __syncthreads();
  }
}

extern "C" void kernel_launch(void* const* d_in, const int* in_sizes, int n_in,
                              void* d_out, int out_size, void* d_ws, size_t ws_size,
                              hipStream_t stream) {
  const float* seq      = (const float*)d_in[0];
  const int*   trans    = (const int*)d_in[1];
  const float* W_comp   = (const float*)d_in[2];
  const float* U_head_w = (const float*)d_in[3];
  const float* U_head_b = (const float*)d_in[4];
  const float* U_cl     = (const float*)d_in[5];
  const float* U_cr     = (const float*)d_in[6];
  const float* W_ih     = (const float*)d_in[7];
  const float* W_hh     = (const float*)d_in[8];
  const float* b_ih     = (const float*)d_in[9];
  const float* b_hh     = (const float*)d_in[10];
  const float* th0      = (const float*)d_in[11];
  const float* tc0      = (const float*)d_in[12];
  char*  ws  = (char*)d_ws;
  float* out = (float*)d_out;

  (void)in_sizes; (void)n_in; (void)out_size; (void)ws_size;
  (void)hipFuncSetAttribute(reinterpret_cast<const void*>(spinn_kernel),
                            hipFuncAttributeMaxDynamicSharedMemorySize, SM_TOTAL);

  pack_kernel<<<1024, 256, 0, stream>>>(W_comp, U_head_w, U_head_b, U_cl, U_cr,
                                        W_ih, W_hh, b_ih, b_hh, ws);
  spinn_kernel<<<Bn / 16, NTHREADS, SM_TOTAL, stream>>>(seq, trans, th0, tc0, ws, out);
}
